// WindowAttention_25761213841771
// MI455X (gfx1250) — matrix operand order repaired and checked by reading, not compiled, not run on hardware
//
#include <hip/hip_runtime.h>
#include <math.h>

typedef __attribute__((ext_vector_type(8)))  unsigned short v8us;
typedef __attribute__((ext_vector_type(16))) unsigned short v16us;
typedef __attribute__((ext_vector_type(16))) __bf16         v16bf;
typedef __attribute__((ext_vector_type(8)))  float          v8f;
typedef __attribute__((ext_vector_type(4)))  int            v4i;

#define SUPPORT 1
#define QUERY   4096
#define CNUM    5
#define DIM     128
#define HEADS   8
#define NTOK    25
#define NWIN    20480          // 5*QUERY
#define D3      384            // 3*DIM
#define PK      3200           // NTOK*DIM
#define PN      6400           // 2*PK

// ---------------- CDNA5 async-load-to-LDS plumbing ----------------
#if defined(__has_builtin)
#  if __has_builtin(__builtin_amdgcn_global_load_async_to_lds_b128)
#    define HAVE_ASYNC_B128 1
#  endif
#  if __has_builtin(__builtin_amdgcn_s_wait_asynccnt)
#    define HAVE_WAIT_ASYNC 1
#  endif
#endif
#ifndef HAVE_ASYNC_B128
#define HAVE_ASYNC_B128 0
#endif
#ifndef HAVE_WAIT_ASYNC
#define HAVE_WAIT_ASYNC 0
#endif

typedef __attribute__((address_space(1))) v4i* gptr_v4i;   // global-AS pointer to int4
typedef __attribute__((address_space(3))) v4i* lptr_v4i;   // LDS-AS pointer to int4

__device__ __forceinline__ void async_ld_b128(const unsigned short* g, unsigned short* l) {
#if HAVE_ASYNC_B128
  __builtin_amdgcn_global_load_async_to_lds_b128(
      (gptr_v4i)(void*)(unsigned short*)g, (lptr_v4i)(void*)l, 0, 0);
#else
  unsigned lds_off = (unsigned)(unsigned long long)(__attribute__((address_space(3))) void*)l;
  asm volatile("global_load_async_to_lds_b128 %0, %1, off"
               :: "v"(lds_off), "v"(g) : "memory");
#endif
}

#if HAVE_WAIT_ASYNC
#define WAIT_ASYNC(n) __builtin_amdgcn_s_wait_asynccnt(n)
#else
#define WAIT_ASYNC(n) asm volatile("s_wait_asynccnt %0" :: "i"(n) : "memory")
#endif

// ---------------- helpers ----------------
__device__ __forceinline__ unsigned short f2bf(float f) {
  union { float f; unsigned int u; } x; x.f = f;
  unsigned int r = x.u + 0x7FFFu + ((x.u >> 16) & 1u);
  return (unsigned short)(r >> 16);
}
__device__ __forceinline__ float bf2f(unsigned short h) {
  union { unsigned int u; float f; } x; x.u = ((unsigned int)h) << 16;
  return x.f;
}
__device__ __forceinline__ v8us zero8() { v8us z = {0,0,0,0,0,0,0,0}; return z; }
__device__ __forceinline__ v8f  vzero8f() { v8f z = {0.f,0.f,0.f,0.f,0.f,0.f,0.f,0.f}; return z; }
__device__ __forceinline__ v16bf mkfrag(v8us lo, v8us hi) {
  union { v16us u; v16bf b; } c;
  c.u = __builtin_shufflevector(lo, hi, 0,1,2,3,4,5,6,7,8,9,10,11,12,13,14,15);
  return c.b;
}
__device__ __forceinline__ v8f wmma_bf16(v16bf a, v16bf b, v8f c) {
  return __builtin_amdgcn_wmma_f32_16x16x32_bf16(false, a, false, b, (short)0, c, false, false);
}

// ---- kernel 0a: qkv_w (128x384 f32, K-major) -> qkvT (384x128 bf16, N-major)
__global__ void k_prep_qkvw(const float* __restrict__ w, unsigned short* __restrict__ wT) {
  int idx = blockIdx.x * 256 + threadIdx.x;
  if (idx < D3 * DIM) {
    int n = idx >> 7, k = idx & 127;
    wT[idx] = f2bf(w[k * D3 + n]);
  }
}

// ---- kernel 0b: proj_w (3200x6400 f32) -> projT (6400x3200 bf16), LDS-tiled transpose
__global__ __launch_bounds__(256) void k_prep_projw(const float* __restrict__ w,
                                                    unsigned short* __restrict__ wT) {
  __shared__ unsigned short t[32][33];
  const int tx = threadIdx.x & 31, ty = threadIdx.x >> 5;  // 32x8
  const int k0 = blockIdx.x * 32, n0 = blockIdx.y * 32;
  for (int r = 0; r < 32; r += 8)
    t[ty + r][tx] = f2bf(w[(size_t)(k0 + ty + r) * PN + n0 + tx]);
  __syncthreads();
  for (int r = 0; r < 32; r += 8)
    wT[(size_t)(n0 + ty + r) * PK + k0 + tx] = t[tx][ty + r];
}

// ---- kernel 0c: out = proj1_b broadcast
__global__ void k_init_out(float* __restrict__ out, const float* __restrict__ p1b) {
  int idx = blockIdx.x * 256 + threadIdx.x;
  if (idx < NWIN * 5) out[idx] = p1b[idx % 5];
}

// ---- kernel A: fused rel-pos + QKV + attention + residual. 1 block per window, 8 waves.
__global__ __launch_bounds__(256) void k_attn(
    const float* __restrict__ x,
    const unsigned short* __restrict__ qkvT,
    const float* __restrict__ qkv_b,
    const float* __restrict__ rel_h,
    const float* __restrict__ rel_w,
    unsigned short* __restrict__ Y)
{
  __shared__ __align__(16) unsigned short xt[32 * 136];        // x + rel, bf16 (pad rows = 0)
  __shared__ __align__(16) unsigned short xo[32 * 128];        // original x (residual), bf16
  __shared__ __align__(16) unsigned short qkv[32 * 392];       // tokens x 384 (q scaled), bf16
  __shared__ __align__(16) unsigned short sP[HEADS * 32 * 32]; // softmax probs, bf16

  const int tid  = threadIdx.x;
  const int lane = tid & 31;
  const int wave = tid >> 5;
  const int llo  = lane & 15;
  const int lhi  = lane >> 4;

  const int win = blockIdx.x;
  const int g   = win / QUERY;
  const int qy  = win - g * QUERY;
  const size_t xbase = ((size_t)(g * (SUPPORT + QUERY) + SUPPORT + qy)) * NTOK * DIM;

  // Phase 1: stage x (+rel) into LDS as bf16
  for (int idx = tid; idx < 32 * DIM; idx += 256) {
    const int t = idx >> 7, c = idx & 127;
    float v = 0.f, vt = 0.f;
    if (t < NTOK) {
      v = x[xbase + (size_t)t * DIM + c];
      const int i = t / CNUM, j = t - i * CNUM;
      const float r = (c < 64) ? rel_h[c * CNUM + i] : rel_w[(c - 64) * CNUM + j];
      vt = v + r;
    }
    xo[t * 128 + c] = f2bf(v);
    xt[t * 136 + c] = f2bf(vt);
  }
  __syncthreads();

  // Phase 2: QKV = xt @ qkv_w + b (M=32pad, N=384, K=128). Each wave: 48 cols.
  for (int nt = 0; nt < 3; ++nt) {
    const int col = wave * 48 + nt * 16 + llo;
    v8f acc0 = vzero8f(), acc1 = vzero8f();
    for (int kt = 0; kt < 4; ++kt) {
      const int kb = kt * 32;
      const unsigned short* bp = qkvT + col * DIM + kb + lhi * 8;    // K chunks +0, +16
      const v16bf bf = mkfrag(*(const v8us*)bp, *(const v8us*)(bp + 16));
      const unsigned short* ap0 = &xt[llo * 136 + kb + lhi * 8];     // K chunks +0, +16
      const v16bf a0 = mkfrag(*(const v8us*)ap0, *(const v8us*)(ap0 + 16));
      const unsigned short* ap1 = ap0 + 16 * 136;
      const v16bf a1 = mkfrag(*(const v8us*)ap1, *(const v8us*)(ap1 + 16));
      acc0 = wmma_bf16(a0, bf, acc0);
      acc1 = wmma_bf16(a1, bf, acc1);
    }
    const float bias = qkv_b[col];
    const float sc = (col < DIM) ? 0.25f : 1.f;  // scale = hd^-0.5 = 1/4 on q
    for (int jj = 0; jj < 8; ++jj) {
      const int m = jj + lhi * 8;
      qkv[m * 392 + col]        = f2bf((acc0[jj] + bias) * sc);
      qkv[(16 + m) * 392 + col] = f2bf((acc1[jj] + bias) * sc);
    }
  }
  __syncthreads();

  // Phase 3: per-head attention; wave == head.
  const int h = wave;
  const int offq = h * 16, offk = DIM + h * 16, offv = 2 * DIM + h * 16;
  unsigned short* P = &sP[h * 1024];

  // S = q k^T  (K=16 padded to 32 with zeros)
  v8f Sa[2][2];
  Sa[0][0] = vzero8f(); Sa[0][1] = vzero8f(); Sa[1][0] = vzero8f(); Sa[1][1] = vzero8f();
  {
    const unsigned short* kp0 = &qkv[llo * 392 + offk];
    v8us kl0 = *(const v8us*)kp0, kh0 = *(const v8us*)(kp0 + 8);
    const unsigned short* kp1 = kp0 + 16 * 392;
    v8us kl1 = *(const v8us*)kp1, kh1 = *(const v8us*)(kp1 + 8);
    if (lhi) { kl0 = zero8(); kh0 = zero8(); kl1 = zero8(); kh1 = zero8(); } // K 16..31 pad
    const v16bf b0 = mkfrag(kl0, kh0);
    const v16bf b1 = mkfrag(kl1, kh1);
    const unsigned short* qp0 = &qkv[llo * 392 + offq];
    const v16bf a0 = mkfrag(*(const v8us*)qp0, *(const v8us*)(qp0 + 8)); // contiguous 16 K; lhi half meets B pad
    const unsigned short* qp1 = qp0 + 16 * 392;
    const v16bf a1 = mkfrag(*(const v8us*)qp1, *(const v8us*)(qp1 + 8));
    Sa[0][0] = wmma_bf16(a0, b0, Sa[0][0]);
    Sa[0][1] = wmma_bf16(a0, b1, Sa[0][1]);
    Sa[1][0] = wmma_bf16(a1, b0, Sa[1][0]);
    Sa[1][1] = wmma_bf16(a1, b1, Sa[1][1]);
  }

  // Register softmax over keys (25 valid of 32), shuffle-reduced across the 16-lane N group.
  const bool n1ok = (16 + llo) < NTOK;
  for (int mt = 0; mt < 2; ++mt) {
    for (int jj = 0; jj < 8; ++jj) {
      const float v0 = Sa[mt][0][jj];
      const float v1m = n1ok ? Sa[mt][1][jj] : -3.0e38f;
      float mx = fmaxf(v0, v1m);
      mx = fmaxf(mx, __shfl_xor(mx, 1, 32));
      mx = fmaxf(mx, __shfl_xor(mx, 2, 32));
      mx = fmaxf(mx, __shfl_xor(mx, 4, 32));
      mx = fmaxf(mx, __shfl_xor(mx, 8, 32));
      const float e0 = __expf(v0 - mx);
      const float e1 = n1ok ? __expf(Sa[mt][1][jj] - mx) : 0.f;
      float sum = e0 + e1;
      sum += __shfl_xor(sum, 1, 32);
      sum += __shfl_xor(sum, 2, 32);
      sum += __shfl_xor(sum, 4, 32);
      sum += __shfl_xor(sum, 8, 32);
      const float inv = 1.f / sum;
      const int m = mt * 16 + jj + lhi * 8;
      P[m * 32 + llo]      = f2bf(e0 * inv);
      P[m * 32 + 16 + llo] = f2bf(e1 * inv);
    }
  }
  asm volatile("" ::: "memory");  // keep LDS store->load order for same-wave RAW

  // O = P @ V  (K=32, keys 25..31 carry zero probability)
  union { v16us u; v16bf b; } vf;
  for (int i = 0; i < 16; ++i) {
    const int kk = (i < 8) ? lhi * 8 + i : 16 + lhi * 8 + (i - 8);
    vf.u[i] = qkv[kk * 392 + offv + llo];
  }
  v8f o0 = vzero8f(), o1 = vzero8f();
  {
    const unsigned short* pp0 = &P[llo * 32 + lhi * 8];
    const v16bf a0 = mkfrag(*(const v8us*)pp0, *(const v8us*)(pp0 + 16));
    const unsigned short* pp1 = pp0 + 16 * 32;
    const v16bf a1 = mkfrag(*(const v8us*)pp1, *(const v8us*)(pp1 + 16));
    o0 = wmma_bf16(a0, vf.b, o0);
    o1 = wmma_bf16(a1, vf.b, o1);
  }

  // residual + store Y (bf16)
  unsigned short* Yw = Y + (size_t)win * PK;
  const int d = h * 16 + llo;
  for (int jj = 0; jj < 8; ++jj) {
    const int m0v = jj + lhi * 8;
    if (m0v < NTOK) Yw[m0v * DIM + d] = f2bf(o0[jj] + bf2f(xo[m0v * 128 + d]));
    const int m1v = 16 + m0v;
    if (m1v < NTOK) Yw[m1v * DIM + d] = f2bf(o1[jj] + bf2f(xo[m1v * 128 + d]));
  }
}

// ---- kernel B: out += (gelu(Y @ proj_w + proj_b)) @ proj1_w.
// Block tile 64(M) x 256(N); 8 waves, each 32x64. K = 3200 in steps of 32.
// Double-buffered LDS staging via GLOBAL_LOAD_ASYNC_TO_LDS_B128 (ASYNCcnt pipeline).
__global__ __launch_bounds__(256) void k_proj(
    const unsigned short* __restrict__ Yg,
    const unsigned short* __restrict__ Wt,    // 6400 x 3200 bf16 (N-major)
    const float* __restrict__ pb,
    const float* __restrict__ p1w,            // 6400 x 5
    float* __restrict__ out)
{
  __shared__ __align__(16) unsigned short Ys[2][64 * 40];
  __shared__ __align__(16) unsigned short Ws[2][256 * 40];
  __shared__ float po[64 * 5];

  const int tid  = threadIdx.x;
  const int lane = tid & 31;
  const int wave = tid >> 5;
  const int llo  = lane & 15;
  const int lhi  = lane >> 4;
  const int wm   = wave >> 2;          // 0..1 -> M half (32 rows)
  const int wn   = wave & 3;           // 0..3 -> N quarter (64 cols)
  const size_t m0 = (size_t)blockIdx.x * 64;
  const int    n0 = blockIdx.y * 256;

  for (int i = tid; i < 64 * 5; i += 256) po[i] = 0.f;

  v8f acc[2][4];
  for (int i = 0; i < 2; ++i)
    for (int j = 0; j < 4; ++j) acc[i][j] = vzero8f();

  const int yrow = tid >> 2, yseg = (tid & 3) * 8;   // Y: 256 b128 chunks, 1/thread

  // stage one K-slab (Y 64x32 + W 256x32) with 5 async b128s per thread
  auto stage = [&](int kt, int buf) {
    const int kb = kt * 32;
    async_ld_b128(Yg + (m0 + yrow) * PK + kb + yseg, &Ys[buf][yrow * 40 + yseg]);
    for (int i = 0; i < 4; ++i) {
      const int c  = tid + 256 * i;                  // W: 1024 b128 chunks, 4/thread
      const int wr = c >> 2, wsg = (c & 3) * 8;
      async_ld_b128(Wt + (size_t)(n0 + wr) * PK + kb + wsg, &Ws[buf][wr * 40 + wsg]);
    }
    // warm L2 two slabs ahead (global_prefetch)
    __builtin_prefetch(Wt + (size_t)(n0 + (tid >> 2) * 4) * PK + kb + 64, 0, 0);
  };

  stage(0, 0);
  for (int kt = 0; kt < 100; ++kt) {
    const int cur = kt & 1;
    if (kt + 1 < 100) {
      stage(kt + 1, cur ^ 1);
      WAIT_ASYNC(5);            // drain slab kt; slab kt+1 stays in flight
    } else {
      WAIT_ASYNC(0);
    }
    __syncthreads();

    const unsigned short* ysb = Ys[cur];
    const unsigned short* wsb = Ws[cur];
    v16bf a0, a1, b[4];
    {
      const unsigned short* p = &ysb[(wm * 32 + llo) * 40 + lhi * 8];
      a0 = mkfrag(*(const v8us*)p, *(const v8us*)(p + 16));
      p += 16 * 40;
      a1 = mkfrag(*(const v8us*)p, *(const v8us*)(p + 16));
    }
    for (int nt = 0; nt < 4; ++nt) {
      const unsigned short* p = &wsb[(wn * 64 + nt * 16 + llo) * 40 + lhi * 8];
      b[nt] = mkfrag(*(const v8us*)p, *(const v8us*)(p + 16));
    }
    for (int nt = 0; nt < 4; ++nt) {
      acc[0][nt] = wmma_bf16(a0, b[nt], acc[0][nt]);
      acc[1][nt] = wmma_bf16(a1, b[nt], acc[1][nt]);
    }
    __syncthreads();            // done reading buf before slab kt+2 overwrites it
  }

  // epilogue: bias + exact GELU + contract against proj1_w into per-block partials
  for (int nt = 0; nt < 4; ++nt) {
    const int ng = n0 + wn * 64 + nt * 16 + llo;
    const float bias = pb[ng];
    float w5[5];
    for (int c = 0; c < 5; ++c) w5[c] = p1w[ng * 5 + c];
    for (int mt = 0; mt < 2; ++mt) {
      for (int jj = 0; jj < 8; ++jj) {
        const int ml = wm * 32 + mt * 16 + jj + lhi * 8;
        const float hv = acc[mt][nt][jj] + bias;
        const float gv = 0.5f * hv * (1.f + erff(hv * 0.70710678118654752f));
        for (int c = 0; c < 5; ++c)
          atomicAdd(&po[ml * 5 + c], gv * w5[c]);
      }
    }
  }
  __syncthreads();
  for (int i = tid; i < 64 * 5; i += 256)
    atomicAdd(&out[(m0 + i / 5) * 5 + i % 5], po[i]);
}

extern "C" void kernel_launch(void* const* d_in, const int* in_sizes, int n_in,
                              void* d_out, int out_size, void* d_ws, size_t ws_size,
                              hipStream_t stream) {
  (void)in_sizes; (void)n_in; (void)out_size; (void)ws_size;
  const float* x      = (const float*)d_in[0];
  const float* qkv_w  = (const float*)d_in[1];
  const float* qkv_b  = (const float*)d_in[2];
  const float* rel_h  = (const float*)d_in[3];
  const float* rel_w  = (const float*)d_in[4];
  const float* proj_w = (const float*)d_in[5];
  const float* proj_b = (const float*)d_in[6];
  const float* p1w    = (const float*)d_in[7];
  const float* p1b    = (const float*)d_in[8];
  float* out = (float*)d_out;

  unsigned short* wsu   = (unsigned short*)d_ws;
  unsigned short* qkvT  = wsu;                              //   384*128 bf16
  unsigned short* projT = wsu + (size_t)D3 * DIM;           //  6400*3200 bf16
  unsigned short* Y     = projT + (size_t)PN * PK;          // 20480*3200 bf16

  k_prep_qkvw<<<(D3 * DIM + 255) / 256, 256, 0, stream>>>(qkv_w, qkvT);
  k_prep_projw<<<dim3(PK / 32, PN / 32), 256, 0, stream>>>(proj_w, projT);
  k_init_out<<<(NWIN * 5 + 255) / 256, 256, 0, stream>>>(out, p1b);
  k_attn<<<NWIN, 256, 0, stream>>>(x, qkvT, qkv_b, rel_h, rel_w, Y);
  k_proj<<<dim3(NWIN / 64, PN / 256), 256, 0, stream>>>(Y, projT, proj_b, p1w, out);
}
